// wAttnModule_38886633898207
// MI455X (gfx1250) — matrix operand order repaired and checked by reading, not compiled, not run on hardware
//
#include <hip/hip_runtime.h>
#include <hip/hip_bf16.h>

#define F   50
#define FP  64      // F padded to 64 (zero-filled) for WMMA K
#define S   256
#define Q   128
#define NDOC 1024

typedef __attribute__((ext_vector_type(16))) __bf16 v16bf;
typedef __attribute__((ext_vector_type(8)))  float  v8f;
typedef int i32x4 __attribute__((vector_size(16)));

#if __has_builtin(__builtin_amdgcn_global_load_async_to_lds_b128)
#define HAVE_ASYNC_LDS 1
#endif

// ---------------- WMMA fragment helpers (CDNA5 wave32 layouts) ----------------
// A (16x32 bf16): lane l, half=l>>4, m=l&15. Elements 0..15 hold K=16*half+{0..15};
// one contiguous 32B chunk per lane (matches ldB_tr).
__device__ __forceinline__ v16bf ldA(const __bf16* src, int ld, int mb, int kb) {
  int lane = threadIdx.x & 31;
  int half = lane >> 4, l16 = lane & 15;
  const __bf16* p = src + (size_t)(mb + l16) * ld + kb;
  v16bf a;
#pragma unroll
  for (int e = 0; e < 16; ++e) a[e] = p[16 * half + e];
  return a;
}

// B fragment where logical B = transpose(src): B[k,n] = src[n*ld + k].
// Contiguous 32B per lane -> ds_load_b128 pair.
__device__ __forceinline__ v16bf ldB_tr(const __bf16* src, int ld, int kb, int nb) {
  int lane = threadIdx.x & 31;
  int half = lane >> 4, l16 = lane & 15;
  const __bf16* p = src + (size_t)(nb + l16) * ld + kb + 16 * half;
  v16bf b;
#pragma unroll
  for (int e = 0; e < 16; ++e) b[e] = p[e];
  return b;
}

// C/D f32 (16x16): VGPR v holds row mb+v+8*half, col nb+(lane&15)
__device__ __forceinline__ void stC_bf16(__bf16* dst, int ld, int mb, int nb, v8f c) {
  int lane = threadIdx.x & 31;
  int half = lane >> 4, l16 = lane & 15;
#pragma unroll
  for (int v = 0; v < 8; ++v)
    dst[(size_t)(mb + v + 8 * half) * ld + nb + l16] = (__bf16)c[v];
}

__device__ __forceinline__ v8f wmma_bf16(v16bf a, v16bf b, v8f c) {
  return __builtin_amdgcn_wmma_f32_16x16x32_bf16(
      /*neg_a=*/false, a, /*neg_b=*/false, b,
      /*c_mod=*/(short)0, c, /*reuse_a=*/false, /*reuse_b=*/false);
}

// ---------------- per-doc kernel: one workgroup (8 wave32) per doc ----------------
__global__ __launch_bounds__(256) void doc_kernel(
    const float* __restrict__ query, const float* __restrict__ doc,
    const float* __restrict__ cl_w, const float* __restrict__ dl1_w,
    const float* __restrict__ dW1p, const float* __restrict__ dl2_w,
    const float* __restrict__ dW2p, float* __restrict__ xArr,
    float* __restrict__ bArr) {
  __shared__ __bf16 sXY[S * Q];          // 64KB: sXY bf16; later aD2Q|aQ2D
  __shared__ __bf16 sD2Q[S * Q];         // 64KB: dw3B aliases its front
  __shared__ __bf16 sQ2D[S * Q];         // 64KB: f32 doc staging aliases it in P0
  __shared__ __bf16 qB[Q * FP];          // 16KB  query row-major (q x f)
  __shared__ __bf16 qT[FP * Q];          // 16KB  query transposed (f x q)
  __shared__ __bf16 docT[FP * S];        // 32KB  doc transposed (f x s)
  __shared__ __bf16 mScr[8][16 * 64];    // 16KB  per-wave m row-block scratch
  __shared__ float dotw1[S], qdotA[Q];
  __shared__ float rowmax[S], rowsum[S], colmax[Q], colsum[Q];
  __shared__ float cArr[S], xLoc[4 * F], red[4];

  __bf16* dw3B = sD2Q;            // doc*w3 (bf16, s x fp), dead before sD2Q written
  __bf16* aD2Q = sXY;             // 256x64 bf16, after sXY is dead
  __bf16* aQ2D = sXY + S * FP;    // 256x64 bf16
  float*  stage = (float*)(void*)sQ2D;  // 51.2KB f32 doc staging, dead before P2

  const int n = blockIdx.x;
  const int tid = threadIdx.x;
  const int wave = tid >> 5;
  const float* docG = doc + (size_t)n * S * F;

  // ---- P0a: stage doc (f32) into LDS (async copy if available) ----
#if HAVE_ASYNC_LDS
  for (int i = tid; i < (S * F) / 4; i += 256) {
    __builtin_amdgcn_global_load_async_to_lds_b128(
        (__attribute__((address_space(1))) i32x4*)(docG + i * 4),
        (__attribute__((address_space(3))) i32x4*)(stage + i * 4), 0, 0);
  }
#if __has_builtin(__builtin_amdgcn_s_wait_asynccnt)
  __builtin_amdgcn_s_wait_asynccnt(0);
#else
  asm volatile("s_wait_asynccnt 0" ::: "memory");
#endif
#else
  for (int i = tid; i < S * F; i += 256) stage[i] = docG[i];
#endif
  __syncthreads();

  // ---- P0b: build bf16 operand panels + row biases (doc read from LDS stage) ----
  for (int i = tid; i < S * FP; i += 256) {       // dw3B = doc*w3 (s x fp)
    int s = i >> 6, f = i & 63;
    dw3B[i] = (__bf16)((f < F) ? stage[s * F + f] * cl_w[2 * F + f] : 0.f);
  }
  for (int i = tid; i < FP * S; i += 256) {       // docT (f x s)
    int f = i >> 8, s = i & 255;
    docT[i] = (__bf16)((f < F) ? stage[s * F + f] : 0.f);
  }
  for (int i = tid; i < Q * FP; i += 256) {       // qB (q x f)
    int q = i >> 6, f = i & 63;
    qB[i] = (__bf16)((f < F) ? query[q * F + f] : 0.f);
  }
  for (int i = tid; i < FP * Q; i += 256) {       // qT (f x q)
    int f = i >> 7, q = i & 127;
    qT[i] = (__bf16)((f < F) ? query[q * F + f] : 0.f);
  }
  {
    int s = tid;
    float acc = 0.f;
    for (int f = 0; f < F; ++f) acc += stage[s * F + f] * cl_w[f];
    dotw1[s] = acc;
  }
  if (tid < Q) {
    float acc = 0.f;
    for (int f = 0; f < F; ++f) acc += query[tid * F + f] * cl_w[F + f];
    qdotA[tid] = acc;
  }
  __syncthreads();

  // ---- P1: sXY = (doc*w3) @ query^T + dotw1[s] + qdot[q]  (WMMA K=64) ----
  for (int si = wave; si < 16; si += 8) {
    v16bf a0 = ldA(dw3B, FP, si * 16, 0);
    v16bf a1 = ldA(dw3B, FP, si * 16, 32);
    for (int qi = 0; qi < 8; ++qi) {
      v8f acc = {};
      acc = wmma_bf16(a0, ldB_tr(qB, FP, 0, qi * 16), acc);   // B = query^T
      acc = wmma_bf16(a1, ldB_tr(qB, FP, 32, qi * 16), acc);
      int lane = tid & 31, half = lane >> 4, l16 = lane & 15;
#pragma unroll
      for (int v = 0; v < 8; ++v) {
        int s = si * 16 + v + 8 * half;
        int q = qi * 16 + l16;
        sXY[s * Q + q] = (__bf16)(acc[v] + dotw1[s] + qdotA[q]);
      }
    }
  }
  __syncthreads();

  // ---- P2: dual softmax (axis=1 -> sD2Q, axis=2 -> sQ2D) ----
  {
    int s = tid;
    float m = -1e30f;
    for (int q = 0; q < Q; ++q) m = fmaxf(m, (float)sXY[s * Q + q]);
    float sum = 0.f;
    for (int q = 0; q < Q; ++q) sum += __expf((float)sXY[s * Q + q] - m);
    rowmax[s] = m; rowsum[s] = sum;
  }
  if (tid < Q) {
    int q = tid;
    float m = -1e30f;
    for (int s = 0; s < S; ++s) m = fmaxf(m, (float)sXY[s * Q + q]);
    float sum = 0.f;
    for (int s = 0; s < S; ++s) sum += __expf((float)sXY[s * Q + q] - m);
    colmax[q] = m; colsum[q] = sum;
  }
  __syncthreads();   // NB: writing sD2Q/sQ2D below also kills dw3B and stage
  {
    int s = tid;
    float rm = rowmax[s], rs = rowsum[s];
    for (int q = 0; q < Q; ++q) {
      float x = (float)sXY[s * Q + q];
      sD2Q[s * Q + q] = (__bf16)(__expf(x - colmax[q]) / colsum[q]);
      sQ2D[s * Q + q] = (__bf16)(__expf(x - rm) / rs);
    }
  }
  __syncthreads();

  // ---- P3: aD2Q = sD2Q @ query   (WMMA K=128, A hoisted per si) ----
  for (int si = wave; si < 16; si += 8) {
    v16bf a[4];
#pragma unroll
    for (int kc = 0; kc < 4; ++kc) a[kc] = ldA(sD2Q, Q, si * 16, kc * 32);
#pragma unroll
    for (int fi = 0; fi < 4; ++fi) {
      v8f acc = {};
#pragma unroll
      for (int kc = 0; kc < 4; ++kc)
        acc = wmma_bf16(a[kc], ldB_tr(qT, Q, kc * 32, fi * 16), acc);
      stC_bf16(aD2Q, FP, si * 16, fi * 16, acc);
    }
  }
  __syncthreads();

  // ---- P4: m = sD2Q @ sQ2D^T (tile-resident), aQ2D = m @ doc  (WMMA) ----
  for (int si = wave; si < 16; si += 8) {
    v16bf aSD[4];
#pragma unroll
    for (int kc = 0; kc < 4; ++kc) aSD[kc] = ldA(sD2Q, Q, si * 16, kc * 32);
    v8f accF[4] = {};
    __bf16* mS = &mScr[wave][0];                   // wave-private 16x64
    for (int tp = 0; tp < 4; ++tp) {               // t in [tp*64, tp*64+64)
#pragma unroll
      for (int ti = 0; ti < 4; ++ti) {
        v8f mc = {};
#pragma unroll
        for (int kc = 0; kc < 4; ++kc)
          mc = wmma_bf16(aSD[kc], ldB_tr(sQ2D, Q, kc * 32, tp * 64 + ti * 16), mc);
        stC_bf16(mS, 64, 0, ti * 16, mc);          // wave-private, no barrier
      }
#pragma unroll
      for (int kc2 = 0; kc2 < 2; ++kc2) {
        v16bf am = ldA(mS, 64, 0, kc2 * 32);       // hoisted across fi
#pragma unroll
        for (int fi = 0; fi < 4; ++fi)
          accF[fi] = wmma_bf16(am, ldB_tr(docT, S, tp * 64 + kc2 * 32, fi * 16),
                               accF[fi]);
      }
    }
#pragma unroll
    for (int fi = 0; fi < 4; ++fi)
      stC_bf16(aQ2D, FP, si * 16, fi * 16, accF[fi]);
  }
  __syncthreads();

  // ---- P5: c[s] = sum_k tanh(dW1*V[s,k])*dl1_w[k] ----
  {
    int s = tid;
    const float* dr = docG + s * F;
    __builtin_prefetch(dr, 0, 0);                  // global_prefetch_b8
    float dw1 = dW1p[0];
    float acc = 0.f;
    for (int j = 0; j < F; ++j) {
      float dv = dr[j];
      float ad = (float)aD2Q[s * FP + j];
      float aq = (float)aQ2D[s * FP + j];
      acc += tanhf(dw1 * dv) * dl1_w[j];
      acc += tanhf(dw1 * ad) * dl1_w[F + j];
      acc += tanhf(dw1 * dv * ad) * dl1_w[2 * F + j];
      acc += tanhf(dw1 * dv * aq) * dl1_w[3 * F + j];
    }
    cArr[s] = acc;
  }
  __syncthreads();
  if (tid == 0) {
    float m = -1e30f;
    for (int s = 0; s < S; ++s) m = fmaxf(m, cArr[s]);
    float sum = 0.f;
    for (int s = 0; s < S; ++s) sum += __expf(cArr[s] - m);
    red[0] = m; red[1] = sum;
  }
  __syncthreads();

  // ---- P6: x = alpha @ V ; b[n] = sum tanh(dW2*x)*dl2_w ----
  if (tid < 4 * F) {
    int k = tid, p = k / F, j = k % F;
    float m = red[0], sum = red[1];
    float acc = 0.f;
    for (int s = 0; s < S; ++s) {
      float alpha = __expf(cArr[s] - m) / sum;
      float dv = docG[s * F + j];
      float v;
      if (p == 0)      v = dv;
      else if (p == 1) v = (float)aD2Q[s * FP + j];
      else if (p == 2) v = dv * (float)aD2Q[s * FP + j];
      else             v = dv * (float)aQ2D[s * FP + j];
      acc += alpha * v;
    }
    xLoc[k] = acc;
    xArr[(size_t)n * (4 * F) + k] = acc;
  }
  __syncthreads();
  if (tid == 0) {
    float dw2 = dW2p[0];
    float b = 0.f;
    for (int k = 0; k < 4 * F; ++k) b += tanhf(dw2 * xLoc[k]) * dl2_w[k];
    bArr[n] = b;
  }
}

// ---------------- final reduction kernel: softmax over docs + head ----------------
__global__ __launch_bounds__(256) void final_kernel(
    const float* __restrict__ query, const float* __restrict__ ql_w,
    const float* __restrict__ qWp, const float* __restrict__ l1_w,
    const float* __restrict__ l1_b, const float* __restrict__ l2_w,
    const float* __restrict__ l2_b, const float* __restrict__ xArr,
    const float* __restrict__ bArr, float* __restrict__ out) {
  __shared__ float yS[4 * F], cqS[Q], zS[F], hS[F], red[4];
  int tid = threadIdx.x;

  if (tid == 0) {
    float m = -1e30f;
    for (int n = 0; n < NDOC; ++n) m = fmaxf(m, bArr[n]);
    float sum = 0.f;
    for (int n = 0; n < NDOC; ++n) sum += __expf(bArr[n] - m);
    red[0] = m; red[1] = sum;
  }
  __syncthreads();
  if (tid < 4 * F) {  // y = beta @ x
    float m = red[0], sum = red[1];
    float acc = 0.f;
    for (int n = 0; n < NDOC; ++n)
      acc += (__expf(bArr[n] - m) / sum) * xArr[(size_t)n * (4 * F) + tid];
    yS[tid] = acc;
  }
  if (tid < Q) {  // cq
    float qw = qWp[0];
    float acc = 0.f;
    for (int f = 0; f < F; ++f) acc += tanhf(qw * query[tid * F + f]) * ql_w[f];
    cqS[tid] = acc;
  }
  __syncthreads();
  if (tid == 0) {
    float m = -1e30f;
    for (int q = 0; q < Q; ++q) m = fmaxf(m, cqS[q]);
    float sum = 0.f;
    for (int q = 0; q < Q; ++q) sum += __expf(cqS[q] - m);
    red[2] = m; red[3] = sum;
  }
  __syncthreads();
  if (tid < F) {  // z = alq @ query
    float m = red[2], sum = red[3];
    float acc = 0.f;
    for (int q = 0; q < Q; ++q)
      acc += (__expf(cqS[q] - m) / sum) * query[q * F + tid];
    zS[tid] = acc;
  }
  __syncthreads();
  if (tid < F) {  // h = (l1_w @ y + l1_b) * z
    float acc = l1_b[tid];
    for (int k = 0; k < 4 * F; ++k) acc += l1_w[tid * (4 * F) + k] * yS[k];
    hS[tid] = acc * zS[tid];
  }
  __syncthreads();
  if (tid == 0) {
    float acc = l2_b[0];
    for (int i = 0; i < F; ++i) acc += l2_w[i] * hS[i];
    out[0] = acc;
  }
}

extern "C" void kernel_launch(void* const* d_in, const int* in_sizes, int n_in,
                              void* d_out, int out_size, void* d_ws, size_t ws_size,
                              hipStream_t stream) {
  (void)in_sizes; (void)n_in; (void)out_size; (void)ws_size;
  const float* query = (const float*)d_in[0];
  const float* doc   = (const float*)d_in[1];
  const float* cl_w  = (const float*)d_in[2];
  const float* dl1_w = (const float*)d_in[3];
  const float* dW1   = (const float*)d_in[4];
  const float* dl2_w = (const float*)d_in[5];
  const float* dW2   = (const float*)d_in[6];
  const float* ql_w  = (const float*)d_in[7];
  const float* qW    = (const float*)d_in[8];
  const float* l1_w  = (const float*)d_in[9];
  const float* l1_b  = (const float*)d_in[10];
  const float* l2_w  = (const float*)d_in[11];
  const float* l2_b  = (const float*)d_in[12];

  float* xArr = (float*)d_ws;                 // 1024 x 200 f32
  float* bArr = xArr + (size_t)NDOC * 4 * F;  // 1024 f32

  doc_kernel<<<NDOC, 256, 0, stream>>>(query, doc, cl_w, dl1_w, dW1, dl2_w, dW2,
                                       xArr, bArr);
  final_kernel<<<1, 256, 0, stream>>>(query, ql_w, qW, l1_w, l1_b, l2_w, l2_b,
                                      xArr, bArr, (float*)d_out);
}
